// RbcNetwork_75797582840801
// MI455X (gfx1250) — matrix operand order repaired and checked by reading, not compiled, not run on hardware
//
#include <hip/hip_runtime.h>

#define NN 96
#define PP (NN * NN)        // 9216 matrices
#define PI_ITERS 50

typedef __attribute__((ext_vector_type(16))) __bf16 v16bf;
typedef __attribute__((ext_vector_type(8)))  float  v8f;

static __device__ __forceinline__ float sigmoidf_fast(float x) {
    return 1.0f / (1.0f + __expf(-x));
}

// One block per matrix p (p = i*96 + j). 192 threads = 6 waves, wave w owns
// output rows [16w, 16w+16). A-fragments (bf16 hi/lo split) are built once in
// registers from global memory (fused sigmoid(wr)*rz + rc); the 50 power
// iterations run entirely from registers + ~2KB LDS (ping-pong b vector).
//
// Normalization is deferred one iteration (result is scale-invariant), so each
// iteration needs exactly one barrier and one cross-lane shuffle.
__launch_bounds__(192)
__global__ void rbc_power_kernel(const float* __restrict__ weights_r,
                                 const float* __restrict__ r_zeros,
                                 const float* __restrict__ r_const,
                                 const float* __restrict__ weights_t,
                                 float* __restrict__ ws)
{
    __shared__ float  b_f32[2][NN];
    __shared__ __align__(32) __bf16 b_hi[2][NN];
    __shared__ __align__(32) __bf16 b_lo[2][NN];
    __shared__ float  partials[2][6];   // per-wave ||v||^2 partials, ping-pong

    const int tid  = threadIdx.x;
    const int lane = tid & 31;
    const int wave = tid >> 5;           // 0..5  (row block)
    const int mrow = lane & 15;          // A-frag row within block
    const int half = (lane >> 4) & 1;    // half-wave (K interleave / row half)

    const int p = blockIdx.x;
    const size_t mbase = (size_t)p * (size_t)PP;

    // ---------- build WMMA A-fragments (iteration-invariant) ----------
    // A 16x32 frag, contiguous K order (B matches): lanes 0-15 rows M=0..15, lanes 16-31 same rows;
    // halfword h -> K = half*16 + h  (16 consecutive K per half-wave)
    v16bf a_hi[3], a_lo[3];
    {
        const int row = wave * 16 + mrow;
        const size_t rbase = mbase + (size_t)row * NN;
        #pragma unroll
        for (int kb = 0; kb < 3; ++kb) {
            const int c0 = kb * 32 + half * 16;
            #pragma unroll
            for (int seg = 0; seg < 2; ++seg) {
                const int c = c0 + seg * 8;
                float wrv[8], rzv[8], rcv[8];
                *(float4*)&wrv[0] = *(const float4*)(weights_r + rbase + c);
                *(float4*)&wrv[4] = *(const float4*)(weights_r + rbase + c + 4);
                *(float4*)&rzv[0] = *(const float4*)(r_zeros  + rbase + c);
                *(float4*)&rzv[4] = *(const float4*)(r_zeros  + rbase + c + 4);
                *(float4*)&rcv[0] = *(const float4*)(r_const  + rbase + c);
                *(float4*)&rcv[4] = *(const float4*)(r_const  + rbase + c + 4);
                #pragma unroll
                for (int e = 0; e < 8; ++e) {
                    float a = sigmoidf_fast(wrv[e]) * rzv[e] + rcv[e];
                    __bf16 h = (__bf16)a;
                    a_hi[kb][seg * 8 + e] = h;
                    a_lo[kb][seg * 8 + e] = (__bf16)(a - (float)h);
                }
            }
        }
    }

    // ---------- v_0 = ones, ||v_0||^2 = 96 (16 per wave) ----------
    if (tid < NN) {
        b_f32[0][tid] = 1.0f;
        b_hi[0][tid]  = (__bf16)1.0f;
        b_lo[0][tid]  = (__bf16)0.0f;
    }
    if (tid < 6) partials[0][tid] = 16.0f;

    // ---------- power iterations (one barrier each) ----------
    for (int it = 0; it < PI_ITERS; ++it) {
        const int cur = it & 1;
        const int nxt = cur ^ 1;

        __syncthreads();   // v_cur + its partials visible; prev-buf reads done

        // inv normalizes the vector currently in buf[cur]
        const float tot = partials[cur][0] + partials[cur][1] + partials[cur][2]
                        + partials[cur][3] + partials[cur][4] + partials[cur][5];
        const float inv = rsqrtf(tot);   // exact scale cancels in final ratio

        // B(32x16): all columns replicate b -> each half-wave loads the same
        // 16 consecutive bf16 values (wide ds loads, bank-broadcast)
        v8f acc0 = {}; v8f acc1 = {}; v8f acc2 = {};
        #pragma unroll
        for (int kb = 0; kb < 3; ++kb) {
            const int off = kb * 32 + half * 16;
            v16bf bh = *(const v16bf*)&b_hi[cur][off];
            v16bf bl = *(const v16bf*)&b_lo[cur][off];
            // v = (Ahi+Alo)(bhi+blo) dropping lo*lo: ~f32-accurate at bf16 rate
            acc0 = __builtin_amdgcn_wmma_f32_16x16x32_bf16(
                       false, a_hi[kb], false, bh, (short)0, acc0, false, false);
            acc1 = __builtin_amdgcn_wmma_f32_16x16x32_bf16(
                       false, a_hi[kb], false, bl, (short)0, acc1, false, false);
            acc2 = __builtin_amdgcn_wmma_f32_16x16x32_bf16(
                       false, a_lo[kb], false, bh, (short)0, acc2, false, false);
        }

        // d = A * (v_cur * inv) = v_next  (unnormalized; |d| ~ lambda, bf16-safe)
        float d[8];
        float s = 0.0f;
        #pragma unroll
        for (int r = 0; r < 8; ++r) {
            d[r] = (acc0[r] + acc1[r] + acc2[r]) * inv;
            s += d[r] * d[r];
        }
        // columns are replicated: lanes<16 all hold the low-8 rows, lanes>=16
        // the high-8 rows -> one xor-16 shuffle gives the 16-row partial
        s += __shfl_xor(s, 16, 32);
        if (lane == 0) partials[nxt][wave] = s;

        if ((lane & 15) == 0) {   // D column N==0 lanes: rows 16w + half*8 + r
            const int rowbase = wave * 16 + half * 8;
            #pragma unroll
            for (int r = 0; r < 8; ++r) {
                const float nb = d[r];
                b_f32[nxt][rowbase + r] = nb;
                __bf16 h = (__bf16)nb;
                b_hi[nxt][rowbase + r] = h;
                b_lo[nxt][rowbase + r] = (__bf16)(nb - (float)h);
            }
        }
    }
    __syncthreads();

    // ---------- epilogue: n_ev contribution (scale-invariant ratio) ----------
    const int fin = PI_ITERS & 1;               // buffer holding v_50
    const int i = p / NN;                       // src_vals: ev[i,j,i] = b[i]
    const float scale = weights_t[p] / b_f32[fin][i];
    if (tid < NN)
        ws[(size_t)tid * PP + p] = b_f32[fin][tid] * scale;
}

// out[k] = sum_p ws[k*9216 + p]  (contiguous, deterministic tree reduction)
__global__ void rbc_reduce_kernel(const float* __restrict__ ws,
                                  float* __restrict__ out)
{
    __shared__ float sm[256];
    const int k = blockIdx.x;
    const float* src = ws + (size_t)k * PP;
    float s = 0.0f;
    for (int pidx = threadIdx.x; pidx < PP; pidx += 256)
        s += src[pidx];
    sm[threadIdx.x] = s;
    __syncthreads();
    for (int m = 128; m > 0; m >>= 1) {
        if (threadIdx.x < m) sm[threadIdx.x] += sm[threadIdx.x + m];
        __syncthreads();
    }
    if (threadIdx.x == 0) out[k] = sm[0];
}

extern "C" void kernel_launch(void* const* d_in, const int* in_sizes, int n_in,
                              void* d_out, int out_size, void* d_ws, size_t ws_size,
                              hipStream_t stream) {
    (void)in_sizes; (void)n_in; (void)out_size; (void)ws_size;
    const float* weights_t = (const float*)d_in[0];
    const float* weights_r = (const float*)d_in[1];
    const float* r_zeros   = (const float*)d_in[2];
    const float* r_const   = (const float*)d_in[3];
    float* out = (float*)d_out;
    float* ws  = (float*)d_ws;   // needs 96*9216*4 = 3.54 MB; fully rewritten each call

    rbc_power_kernel<<<dim3(PP), dim3(192), 0, stream>>>(
        weights_r, r_zeros, r_const, weights_t, ws);
    rbc_reduce_kernel<<<dim3(NN), dim3(256), 0, stream>>>(ws, out);
}
